// MaskTransformer_33054068310159
// MI455X (gfx1250) — matrix operand order repaired and checked by reading, not compiled, not run on hardware
//
#include <hip/hip_runtime.h>

typedef _Float16 f16;
typedef __attribute__((ext_vector_type(16))) _Float16 v16h;
typedef __attribute__((ext_vector_type(8)))  _Float16 v8h;
typedef __attribute__((ext_vector_type(8)))  float    v8f;

#define DMODEL 768
#define SP     1056   // padded sequence (1043 -> 66*16)
#define SEQ    1043
#define NCLS   19
#define NBATCH 8
#define NHEAD  12
#define DHEAD  64
#define MLPD   3072
#define MTOT   (NBATCH*SP)   // 8448

// ---------------------------------------------------------------- WMMA helpers

__device__ __forceinline__ v8f wmma32(v16h a, v16h b, v8f c) {
  return __builtin_amdgcn_wmma_f32_16x16x32_f16(false, a, false, b, (short)0, c,
                                                false, false);
}

// A-fragment (16x32 f16, row-major source):
// lane l: m = l&15, h = l>>4 ; element i -> K = 16h + i
// (two adjacent 16B chunks)
__device__ __forceinline__ v16h load_a_frag(const f16* base, long lda,
                                            int ml, int hl, int k0) {
  const f16* p = base + (long)ml * lda + k0 + 16 * hl;
  v8h x0 = *(const v8h*)p;
  v8h x1 = *(const v8h*)(p + 8);
  return __builtin_shufflevector(x0, x1, 0,1,2,3,4,5,6,7,8,9,10,11,12,13,14,15);
}
// B-fragment (32x16): lane l: n = l&15, h = l>>4 ; element i -> K = 16h + i
// With W stored transposed [N][K], that is ONE contiguous v16h load.

__device__ __forceinline__ float gelu_exact(float v) {
  return 0.5f * v * (1.0f + erff(v * 0.70710678118654752f));
}

// ---------------------------------------------------------------- generic GEMM
// C[M,N] = act(A[M,K] @ Bt[N,K]^T + bias); per-wave 32x64 tile:
// 2 A-fragments share 4 B-fragments -> 8 WMMAs per 12 b128 loads.
__global__ __launch_bounds__(128)
void gemm_kernel(const f16* __restrict__ A, long lda, long aStrideZ,
                 const f16* __restrict__ Bt, long ldb, long bStrideZ,
                 const float* __restrict__ bias,
                 float* __restrict__ Cf, long ldc, long cStrideZ, int addRes,
                 f16* __restrict__ Ch, long ldch, long chStrideZ,
                 int M, int N, int K, int act) {
  int wave = threadIdx.x >> 5, lane = threadIdx.x & 31;
  int mt = blockIdx.x * 4 + wave;
  if (mt * 32 >= M) return;                       // wave-uniform
  int n0 = blockIdx.y * 64;
  long z = blockIdx.z;
  int ml = lane & 15, hl = lane >> 4;
  const f16* Az = A + z * aStrideZ + (long)mt * 32 * lda;
  const f16* Bz = Bt + z * bStrideZ + (long)n0 * ldb;
  v8f acc[2][4] = {};
  for (int k0 = 0; k0 < K; k0 += 32) {
    v16h a0 = load_a_frag(Az, lda, ml, hl, k0);
    v16h a1 = load_a_frag(Az + 16 * lda, lda, ml, hl, k0);
#pragma unroll
    for (int t = 0; t < 4; t++) {
      v16h b = *(const v16h*)(Bz + (long)(t * 16 + ml) * ldb + k0 + 16 * hl);
      acc[0][t] = wmma32(a0, b, acc[0][t]);
      acc[1][t] = wmma32(a1, b, acc[1][t]);
    }
  }
#pragma unroll
  for (int half = 0; half < 2; half++) {
#pragma unroll
    for (int t = 0; t < 4; t++) {
      int n = n0 + t * 16 + ml;
      if (n >= N) continue;
#pragma unroll
      for (int r = 0; r < 8; r++) {
        int m = mt * 32 + half * 16 + r + 8 * hl;
        if (m >= M) continue;
        float v = acc[half][t][r];
        if (bias) v += bias[n];
        if (act) v = gelu_exact(v);
        if (Cf) {
          float* p = Cf + z * cStrideZ + (long)m * ldc + n;
          *p = addRes ? (*p + v) : v;
        }
        if (Ch) Ch[z * chStrideZ + (long)m * ldch + n] = (f16)v;
      }
    }
  }
}

// ---------------------------------------------------- per-head QKV projection
// out[s, h*64+e] = sum_d H[s, h*64+d] * Wt[h][e][d] + bias[h*64+e]
// per-wave 32x64 tile (full head width), 33 row-tiles cover SP=1056.
__global__ __launch_bounds__(128)
void headproj_kernel(const f16* __restrict__ Hbuf, const f16* __restrict__ Wt,
                     const float* __restrict__ bias,
                     f16* __restrict__ outRM, f16* __restrict__ outT) {
  int wave = threadIdx.x >> 5, lane = threadIdx.x & 31;
  int mt = blockIdx.x * 4 + wave;
  if (mt >= SP / 32) return;
  int h = blockIdx.y;
  long b = blockIdx.z;
  int ml = lane & 15, hl = lane >> 4;
  const f16* Az = Hbuf + (b * SP + (long)mt * 32) * DMODEL + h * DHEAD;
  const f16* Bz = Wt + (long)h * DHEAD * DHEAD;
  v8f acc[2][4] = {};
#pragma unroll
  for (int k0 = 0; k0 < DHEAD; k0 += 32) {
    v16h a0 = load_a_frag(Az, DMODEL, ml, hl, k0);
    v16h a1 = load_a_frag(Az + 16 * DMODEL, DMODEL, ml, hl, k0);
#pragma unroll
    for (int t = 0; t < 4; t++) {
      v16h bf = *(const v16h*)(Bz + (long)(t * 16 + ml) * DHEAD + k0 + 16 * hl);
      acc[0][t] = wmma32(a0, bf, acc[0][t]);
      acc[1][t] = wmma32(a1, bf, acc[1][t]);
    }
  }
#pragma unroll
  for (int half = 0; half < 2; half++) {
#pragma unroll
    for (int t = 0; t < 4; t++) {
      int n = t * 16 + ml;
      float bv = bias[h * DHEAD + n];
#pragma unroll
      for (int r = 0; r < 8; r++) {
        int m = mt * 32 + half * 16 + r + 8 * hl;
        float v = acc[half][t][r] + bv;
        if (outRM) outRM[(b * SP + m) * (long)DMODEL + h * DHEAD + n] = (f16)v;
        else       outT[((b * NHEAD + h) * (long)DHEAD + n) * SP + m] = (f16)v;
      }
    }
  }
}

// --------------------------------------------------------------- fused attention
// one block per (qtile, head, batch): scores -> softmax -> o, x += o
__global__ __launch_bounds__(128)
void attn_kernel(const f16* __restrict__ Q, const f16* __restrict__ Kf,
                 const f16* __restrict__ Vt, float* __restrict__ X) {
  __shared__ __align__(16) f16 sc[16 * SP];
  __shared__ float sinv[16];
  int qt = blockIdx.x, h = blockIdx.y;
  long b = blockIdx.z;
  int wave = threadIdx.x >> 5, lane = threadIdx.x & 31;
  int ml = lane & 15, hl = lane >> 4;

  // phase 1: scores = (Q K^T) / 8, f16 into LDS, pad keys masked
  const f16* Qb = Q + (b * SP + (long)qt * 16) * DMODEL + h * DHEAD;
  v16h a0 = load_a_frag(Qb, DMODEL, ml, hl, 0);
  v16h a1 = load_a_frag(Qb, DMODEL, ml, hl, 32);
  for (int nt = wave; nt < SP / 16; nt += 4) {
    const f16* Kb = Kf + (b * SP + (long)nt * 16) * DMODEL + h * DHEAD;
    v16h b0 = *(const v16h*)(Kb + (long)ml * DMODEL + 16 * hl);
    v16h b1 = *(const v16h*)(Kb + (long)ml * DMODEL + 32 + 16 * hl);
    v8f c = {};
    c = wmma32(a0, b0, c);
    c = wmma32(a1, b1, c);
    int key = nt * 16 + ml;
#pragma unroll
    for (int r = 0; r < 8; r++) {
      float v = (key < SEQ) ? c[r] * 0.125f : -60000.0f;
      sc[(r + 8 * hl) * SP + key] = (f16)v;
    }
  }
  __syncthreads();

  // phase 2: rowwise softmax (store unnormalized exp, keep 1/sum)
  {
    int r = threadIdx.x >> 3, sub = threadIdx.x & 7;
    float mx = -3.0e38f;
    for (int c = sub; c < SP; c += 8) mx = fmaxf(mx, (float)sc[r * SP + c]);
    mx = fmaxf(mx, __shfl_xor(mx, 1));
    mx = fmaxf(mx, __shfl_xor(mx, 2));
    mx = fmaxf(mx, __shfl_xor(mx, 4));
    float s = 0.0f;
    for (int c = sub; c < SP; c += 8) {
      float e = expf((float)sc[r * SP + c] - mx);
      sc[r * SP + c] = (f16)e;
      s += e;
    }
    s += __shfl_xor(s, 1);
    s += __shfl_xor(s, 2);
    s += __shfl_xor(s, 4);
    if (sub == 0) sinv[r] = 1.0f / s;
  }
  __syncthreads();

  // phase 3: o = (exp * V) * (1/sum); each wave owns one 16-wide d-slice
  const f16* Vb = Vt + ((b * NHEAD + h) * (long)DHEAD + wave * 16) * SP;
  v8f acc = {};
  for (int kk = 0; kk < SP; kk += 32) {
    v16h a = load_a_frag(sc, SP, ml, hl, kk);
    v16h bf = *(const v16h*)(Vb + (long)ml * SP + kk + 16 * hl);
    acc = wmma32(a, bf, acc);
  }
#pragma unroll
  for (int r = 0; r < 8; r++) {
    int m = qt * 16 + r + 8 * hl;
    float v = acc[r] * sinv[r + 8 * hl];
    X[(b * SP + m) * (long)DMODEL + h * DHEAD + wave * 16 + ml] += v;
  }
}

// ---------------------------------------------------------------- LayerNorm
__global__ __launch_bounds__(256)
void ln_kernel(const float* __restrict__ X, const float* __restrict__ g,
               const float* __restrict__ bt, f16* __restrict__ out) {
  __shared__ float r1[256], r2[256];
  long row = blockIdx.x;
  int tid = threadIdx.x;
  const float* xr = X + row * DMODEL;
  float s = 0.f, q = 0.f;
  for (int c = tid; c < DMODEL; c += 256) { float v = xr[c]; s += v; q += v * v; }
  r1[tid] = s; r2[tid] = q;
  __syncthreads();
  for (int st = 128; st > 0; st >>= 1) {
    if (tid < st) { r1[tid] += r1[tid + st]; r2[tid] += r2[tid + st]; }
    __syncthreads();
  }
  float mean = r1[0] * (1.0f / DMODEL);
  float var  = r2[0] * (1.0f / DMODEL) - mean * mean;
  float rs = rsqrtf(var + 1e-5f);
  for (int c = tid; c < DMODEL; c += 256)
    out[row * DMODEL + c] = (f16)((xr[c] - mean) * rs * g[c] + bt[c]);
}

// ------------------------------------------------------------- L2 row normalize
__global__ __launch_bounds__(256)
void rownorm_kernel(const float* __restrict__ X, f16* __restrict__ out,
                    int rowsPerBatch, int validRows) {
  __shared__ float r2[256];
  long row = blockIdx.x;
  int tid = threadIdx.x;
  bool valid = ((int)(row % rowsPerBatch)) < validRows;
  const float* xr = X + row * DMODEL;
  float q = 0.f;
  if (valid) for (int c = tid; c < DMODEL; c += 256) { float v = xr[c]; q += v * v; }
  r2[tid] = q;
  __syncthreads();
  for (int st = 128; st > 0; st >>= 1) {
    if (tid < st) r2[tid] += r2[tid + st];
    __syncthreads();
  }
  float rs = rsqrtf(fmaxf(r2[0], 1e-30f));
  for (int c = tid; c < DMODEL; c += 256)
    out[row * DMODEL + c] = (f16)(valid ? xr[c] * rs : 0.0f);
}

// ---------------------------------------------------------- weight transp+cvt
__global__ void transcvt_kernel(const float* __restrict__ in, f16* __restrict__ out,
                                int nb, int K, int N) {
  long i = (long)blockIdx.x * blockDim.x + threadIdx.x;
  long total = (long)nb * K * N;
  if (i >= total) return;
  long b = i / ((long)K * N), rem = i % ((long)K * N);
  long k = rem / N, n = rem % N;
  out[b * (long)K * N + n * (long)K + k] = (f16)in[i];
}

__global__ void cvt_f16_kernel(const float* __restrict__ in, f16* __restrict__ out,
                               long n) {
  long i = (long)blockIdx.x * blockDim.x + threadIdx.x;
  if (i < n) out[i] = (f16)in[i];
}

// cls tokens + zero padding rows
__global__ void clsfill_kernel(const float* __restrict__ cls, float* __restrict__ X) {
  long i = (long)blockIdx.x * blockDim.x + threadIdx.x;
  if (i >= (long)NBATCH * 32 * DMODEL) return;
  long c = i % DMODEL, rr = (i / DMODEL) % 32, b = i / (32L * DMODEL);
  X[(b * SP + 1024 + rr) * (long)DMODEL + c] = (rr < NCLS) ? cls[rr * DMODEL + c] : 0.0f;
}

// LN over classes + layout [b][c][32][32]
__global__ void maskln_kernel(const float* __restrict__ masks,
                              const float* __restrict__ g, const float* __restrict__ bt,
                              float* __restrict__ mln) {
  long i = (long)blockIdx.x * blockDim.x + threadIdx.x;
  if (i >= (long)NBATCH * 1024) return;
  long b = i / 1024, n = i % 1024;
  const float* mr = masks + (b * 1024 + n) * NCLS;
  float s = 0.f;
  for (int c = 0; c < NCLS; c++) s += mr[c];
  float mean = s * (1.0f / NCLS);
  float q = 0.f;
  for (int c = 0; c < NCLS; c++) { float d = mr[c] - mean; q += d * d; }
  float rs = rsqrtf(q * (1.0f / NCLS) + 1e-5f);
  for (int c = 0; c < NCLS; c++)
    mln[(b * NCLS + c) * 1024 + n] = (mr[c] - mean) * rs * g[c] + bt[c];
}

// half-pixel bilinear 32x32 -> 512x512
__global__ void resize_kernel(const float* __restrict__ mln, float* __restrict__ out) {
  long i = (long)blockIdx.x * blockDim.x + threadIdx.x;
  if (i >= (long)NBATCH * NCLS * 512 * 512) return;
  int x = (int)(i % 512);
  long t = i / 512;
  int y = (int)(t % 512); t /= 512;
  int c = (int)(t % NCLS);
  int b = (int)(t / NCLS);
  float fy = (y + 0.5f) * 0.0625f - 0.5f;
  float fx = (x + 0.5f) * 0.0625f - 0.5f;
  int y0 = (int)floorf(fy), x0 = (int)floorf(fx);
  float ty = fy - (float)y0, tx = fx - (float)x0;
  int y1 = y0 + 1 > 31 ? 31 : y0 + 1;
  int x1 = x0 + 1 > 31 ? 31 : x0 + 1;
  y0 = y0 < 0 ? 0 : y0;
  x0 = x0 < 0 ? 0 : x0;
  const float* m = mln + ((long)(b * NCLS + c)) * 1024;
  float v00 = m[y0 * 32 + x0], v01 = m[y0 * 32 + x1];
  float v10 = m[y1 * 32 + x0], v11 = m[y1 * 32 + x1];
  out[i] = (1.f - ty) * ((1.f - tx) * v00 + tx * v01) +
           ty * ((1.f - tx) * v10 + tx * v11);
}

// ================================================================ host driver
extern "C" void kernel_launch(void* const* d_in, const int* in_sizes, int n_in,
                              void* d_out, int out_size, void* d_ws, size_t ws_size,
                              hipStream_t stream) {
  (void)in_sizes; (void)n_in; (void)out_size; (void)ws_size;
  const float* x_in   = (const float*)d_in[0];
  const float* Wdec   = (const float*)d_in[1];
  const float* bdec   = (const float*)d_in[2];
  const float* clsemb = (const float*)d_in[3];
  const float* ln1g   = (const float*)d_in[4];
  const float* ln1b   = (const float*)d_in[5];
  const float* qW     = (const float*)d_in[6];
  const float* qb     = (const float*)d_in[7];
  const float* kW     = (const float*)d_in[8];
  const float* kb     = (const float*)d_in[9];
  const float* vW     = (const float*)d_in[10];
  const float* vb     = (const float*)d_in[11];
  const float* ln2g   = (const float*)d_in[12];
  const float* ln2b   = (const float*)d_in[13];
  const float* W1     = (const float*)d_in[14];
  const float* b1     = (const float*)d_in[15];
  const float* W2     = (const float*)d_in[16];
  const float* b2     = (const float*)d_in[17];
  const float* Wp     = (const float*)d_in[18];
  const float* Wc     = (const float*)d_in[19];
  const float* decg   = (const float*)d_in[20];
  const float* decb   = (const float*)d_in[21];
  const float* mg     = (const float*)d_in[22];
  const float* mb     = (const float*)d_in[23];
  float* out = (float*)d_out;

  char* ws = (char*)d_ws;
  size_t off = 0;
  auto alloc = [&](size_t bytes) -> char* {
    char* p = ws + off;
    off = (off + bytes + 255) & ~(size_t)255;
    return p;
  };
  float* xbuf = (float*)alloc((size_t)NBATCH * SP * DMODEL * 4);
  f16* Hh  = (f16*)alloc((size_t)NBATCH * SP * DMODEL * 2);
  f16* Qh  = (f16*)alloc((size_t)NBATCH * SP * DMODEL * 2);
  f16* Kh  = (f16*)alloc((size_t)NBATCH * SP * DMODEL * 2);
  f16* Vth = (f16*)alloc((size_t)NBATCH * NHEAD * DHEAD * SP * 2);
  f16* Gh  = (f16*)alloc((size_t)MTOT * MLPD * 2);
  f16* x0h = (f16*)alloc((size_t)NBATCH * 1024 * DMODEL * 2);
  f16* wdec = (f16*)alloc((size_t)DMODEL * DMODEL * 2);
  f16* wq   = (f16*)alloc((size_t)2 * NHEAD * DHEAD * DHEAD * 2);
  f16* wk   = (f16*)alloc((size_t)2 * NHEAD * DHEAD * DHEAD * 2);
  f16* wv   = (f16*)alloc((size_t)2 * NHEAD * DHEAD * DHEAD * 2);
  f16* w1t  = (f16*)alloc((size_t)2 * DMODEL * MLPD * 2);
  f16* w2t  = (f16*)alloc((size_t)2 * MLPD * DMODEL * 2);
  f16* wpt  = (f16*)alloc((size_t)DMODEL * DMODEL * 2);
  f16* wct  = (f16*)alloc((size_t)DMODEL * DMODEL * 2);
  // final-stage aliases (previous contents dead by then)
  float* P   = (float*)Gh;                                   // [B][1024][768] f32
  f16*   Pn  = (f16*)((char*)Gh + (size_t)32 * 1024 * 1024); // [B][1024][768] f16
  float* Cls = (float*)Qh;                                   // [B][64][768] f32
  f16*   Csn = (f16*)Kh;                                     // [B][64][768] f16
  float* Msk = (float*)Vth;                                  // [B][1024][19]
  f16*   Dh  = Hh;                                           // dec-LN output
  float* mln = (float*)x0h;                                  // [B][19][32][32]

  auto g1 = [](long n) { return dim3((unsigned)((n + 255) / 256)); };

  // weight prep (transpose + f16)
  transcvt_kernel<<<g1(768L * 768), 256, 0, stream>>>(Wdec, wdec, 1, 768, 768);
  transcvt_kernel<<<g1(24L * 64 * 64), 256, 0, stream>>>(qW, wq, 24, 64, 64);
  transcvt_kernel<<<g1(24L * 64 * 64), 256, 0, stream>>>(kW, wk, 24, 64, 64);
  transcvt_kernel<<<g1(24L * 64 * 64), 256, 0, stream>>>(vW, wv, 24, 64, 64);
  transcvt_kernel<<<g1(2L * 768 * 3072), 256, 0, stream>>>(W1, w1t, 2, 768, 3072);
  transcvt_kernel<<<g1(2L * 3072 * 768), 256, 0, stream>>>(W2, w2t, 2, 3072, 768);
  transcvt_kernel<<<g1(768L * 768), 256, 0, stream>>>(Wp, wpt, 1, 768, 768);
  transcvt_kernel<<<g1(768L * 768), 256, 0, stream>>>(Wc, wct, 1, 768, 768);
  cvt_f16_kernel<<<g1((long)NBATCH * 1024 * 768), 256, 0, stream>>>(
      x_in, x0h, (long)NBATCH * 1024 * 768);

  // x = x @ proj_dec_W + b ; concat cls ; zero pad rows
  gemm_kernel<<<dim3(8, 12, 8), 128, 0, stream>>>(
      x0h, 768, 1024L * 768, wdec, 768, 0, bdec,
      xbuf, 768, (long)SP * 768, 0, nullptr, 0, 0, 1024, 768, 768, 0);
  clsfill_kernel<<<g1((long)NBATCH * 32 * 768), 256, 0, stream>>>(clsemb, xbuf);

  for (int l = 0; l < 2; l++) {
    ln_kernel<<<NBATCH * SP, 256, 0, stream>>>(xbuf, ln1g + l * 768, ln1b + l * 768, Hh);
    headproj_kernel<<<dim3(9, 12, 8), 128, 0, stream>>>(
        Hh, wq + (size_t)l * 12 * 4096, qb + l * 768, Qh, nullptr);
    headproj_kernel<<<dim3(9, 12, 8), 128, 0, stream>>>(
        Hh, wk + (size_t)l * 12 * 4096, kb + l * 768, Kh, nullptr);
    headproj_kernel<<<dim3(9, 12, 8), 128, 0, stream>>>(
        Hh, wv + (size_t)l * 12 * 4096, vb + l * 768, nullptr, Vth);
    attn_kernel<<<dim3(66, 12, 8), 128, 0, stream>>>(Qh, Kh, Vth, xbuf);
    ln_kernel<<<NBATCH * SP, 256, 0, stream>>>(xbuf, ln2g + l * 768, ln2b + l * 768, Hh);
    gemm_kernel<<<dim3(66, 48, 1), 128, 0, stream>>>(
        Hh, 768, 0, w1t + (size_t)l * 3072 * 768, 768, 0, b1 + l * 3072,
        nullptr, 0, 0, 0, Gh, 3072, 0, MTOT, 3072, 768, 1);
    gemm_kernel<<<dim3(66, 12, 1), 128, 0, stream>>>(
        Gh, 3072, 0, w2t + (size_t)l * 768 * 3072, 3072, 0, b2 + l * 768,
        xbuf, 768, 0, 1, nullptr, 0, 0, MTOT, 768, 3072, 0);
  }

  // decoder head
  ln_kernel<<<NBATCH * SP, 256, 0, stream>>>(xbuf, decg, decb, Dh);
  gemm_kernel<<<dim3(8, 12, 8), 128, 0, stream>>>(
      Dh, 768, (long)SP * 768, wpt, 768, 0, nullptr,
      P, 768, 1024L * 768, 0, nullptr, 0, 0, 1024, 768, 768, 0);
  gemm_kernel<<<dim3(1, 12, 8), 128, 0, stream>>>(
      Dh + 1024L * 768, 768, (long)SP * 768, wct, 768, 0, nullptr,
      Cls, 768, 64L * 768, 0, nullptr, 0, 0, NCLS, 768, 768, 0);
  rownorm_kernel<<<NBATCH * 1024, 256, 0, stream>>>(P, Pn, 1 << 30, 1 << 30);
  rownorm_kernel<<<NBATCH * 64, 256, 0, stream>>>(Cls, Csn, 64, NCLS);
  gemm_kernel<<<dim3(8, 1, 8), 128, 0, stream>>>(
      Pn, 768, 1024L * 768, Csn, 768, 64L * 768, nullptr,
      Msk, NCLS, 1024L * NCLS, 0, nullptr, 0, 0, 1024, NCLS, 768, 0);
  maskln_kernel<<<g1((long)NBATCH * 1024), 256, 0, stream>>>(Msk, mg, mb, mln);
  resize_kernel<<<g1((long)NBATCH * NCLS * 512 * 512), 256, 0, stream>>>(mln, out);
}
